// MultiheadFusionAttn_64450279244549
// MI455X (gfx1250) — matrix operand order repaired and checked by reading, not compiled, not run on hardware
//
#include <hip/hip_runtime.h>
#include <hip/hip_bf16.h>

#define DEVI __device__ __forceinline__

// ---- problem constants (match reference) ----
constexpr int Bc = 2, Ec = 1024, Sc = 2048, Hc = 16, Dc = 64;
constexpr int Mc = Bc * Sc;          // 4096 token rows
constexpr float kLamInit = 0.1f;
constexpr float kLnEps   = 1e-5f;

// ---- CDNA5 WMMA types ----
typedef __attribute__((ext_vector_type(16))) __bf16        bf16x16;
typedef __attribute__((ext_vector_type(8)))  float         f32x8;
typedef __attribute__((ext_vector_type(4)))  unsigned int  u32x4;

union FragBF { bf16x16 f; u32x4 u[2]; };

DEVI unsigned short f2bf(float x) {            // fp32 -> bf16 RNE
  unsigned int u = __float_as_uint(x);
  u += 0x7fffu + ((u >> 16) & 1u);
  return (unsigned short)(u >> 16);
}

DEVI f32x8 zero8() { f32x8 z = {0.f,0.f,0.f,0.f,0.f,0.f,0.f,0.f}; return z; }

DEVI f32x8 wmma_bf16(const FragBF& a, const FragBF& b, f32x8 c) {
  // D = A(16x32 bf16) * B(32x16 bf16) + C(16x16 f32)
  return __builtin_amdgcn_wmma_f32_16x16x32_bf16(
      false, a.f, false, b.f, (short)0, c, false, false);
}

// A-operand fragment from row-major [rows, K] (lane row = lr):
//   lanes 0-15 hold K {kb..kb+7, kb+16..kb+23}, lanes 16-31 hold {+8, +24}
DEVI void load_afrag(FragBF& fr, const unsigned short* rowptr, int ahalf) {
  const unsigned short* p = rowptr + ahalf;            // ahalf = 0 or 8
  fr.u[0] = *(const u32x4*)p;
  fr.u[1] = *(const u32x4*)(p + 16);
}
// B-operand fragment from row-major B^T == [N, K] (lane col = lr):
//   lanes 0-15 hold K {kb..kb+7, kb+16..kb+23}, lanes 16-31 hold {+8, +24}
DEVI void load_bfrag(FragBF& fr, const unsigned short* rowptr, int bhalf) {
  const unsigned short* p = rowptr + bhalf;            // bhalf = 0 or 8
  fr.u[0] = *(const u32x4*)p;
  fr.u[1] = *(const u32x4*)(p + 16);
}

// ---- CDNA5 async global->LDS (ASYNCcnt-tracked, no VGPR staging) ----
// flat pointer to a __shared__ object carries the LDS byte offset in its
// low 32 bits (aperture in the high bits), which is exactly what the
// VDST operand of GLOBAL_LOAD_ASYNC_TO_LDS wants.
DEVI void async_load_b128(const void* gptr, void* lptr) {
  unsigned lds_off = (unsigned)(size_t)lptr;
  unsigned long long ga = (unsigned long long)(size_t)gptr;
  asm volatile("global_load_async_to_lds_b128 %0, %1, off"
               :: "v"(lds_off), "v"(ga) : "memory");
}
DEVI void wait_async4() { asm volatile("s_wait_asynccnt 0x4" ::: "memory"); }
DEVI void wait_async0() { asm volatile("s_wait_asynccnt 0x0" ::: "memory"); }

// =====================================================================
// elementwise fp32 -> bf16 cast (weights)
// =====================================================================
__global__ __launch_bounds__(256) void cast_bf16_kernel(
    const float* __restrict__ in, unsigned short* __restrict__ out, int n) {
  for (int i = blockIdx.x * 256 + threadIdx.x; i < n; i += gridDim.x * 256)
    out[i] = f2bf(in[i]);
}

// =====================================================================
// [B,E,S] fp32 -> [B,S,E] bf16 tiled transpose
// =====================================================================
__global__ __launch_bounds__(256) void tcast_kernel(
    const float* __restrict__ in, unsigned short* __restrict__ out) {
  __shared__ float tile[32][33];
  const int b  = blockIdx.z;
  const int e0 = blockIdx.y * 32;
  const int s0 = blockIdx.x * 32;
  const int tx = threadIdx.x;   // 0..31
  const int ty = threadIdx.y;   // 0..7
#pragma unroll
  for (int j = 0; j < 32; j += 8)
    tile[ty + j][tx] = in[((size_t)b * Ec + e0 + ty + j) * Sc + s0 + tx];
  __syncthreads();
#pragma unroll
  for (int j = 0; j < 32; j += 8)
    out[((size_t)b * Sc + s0 + ty + j) * Ec + e0 + tx] = f2bf(tile[tx][ty + j]);
}

// =====================================================================
// bf16 WMMA GEMM: C[M,N] = A[M,K] * W[N,K]^T
// 128x128x32 tiles, double-buffered LDS fed by async global->LDS loads.
// MODE 0: bf16 row-major, *scale            (Q/K projections, Wo)
// MODE 1: bf16 V-transposed [B,H,D,S]       (V projections)
// MODE 2: fp32 transposed out[B,E,S], +bias (W2 final)
// MODE 3: bf16 row-major, relu(v + bias)    (W1)
// =====================================================================
template <int MODE>
__global__ __launch_bounds__(256) void gemm_bf16_kernel(
    const unsigned short* __restrict__ A,
    const unsigned short* __restrict__ Bw,
    const float* __restrict__ bias,
    void* __restrict__ Cout,
    int Mdim, int Ndim, int Kdim, float scale) {
  __shared__ __align__(16) unsigned short As[2][128 * 32];
  __shared__ __align__(16) unsigned short Bs[2][128 * 32];

  const int tid  = threadIdx.x;
  const int lane = tid & 31;
  const int wave = tid >> 5;
  const int bm = blockIdx.y * 128;
  const int bn = blockIdx.x * 128;
  const int wm = (wave >> 2) * 64;    // 2 waves along M
  const int wn = (wave & 3) * 32;     // 4 waves along N
  const int lr    = lane & 15;
  const int ahalf = (lane & 16) ? 8 : 0;
  const int bhalf = (lane & 16) ? 8 : 0;

  // per-thread tile chunks: 512 16B-chunks per 128x32 tile, 2 per thread
  const int row0 = tid >> 2, cc0 = (tid & 3) * 8;            // chunk 0
  const int row1 = (tid + 256) >> 2, cc1 = ((tid + 256) & 3) * 8;

  // issue the 4 async loads (2 A-chunks, 2 B-chunks) for one k-tile
  auto issue = [&](int buf, int k0) {
    async_load_b128(&A[(size_t)(bm + row0) * Kdim + k0 + cc0], &As[buf][row0 * 32 + cc0]);
    async_load_b128(&A[(size_t)(bm + row1) * Kdim + k0 + cc1], &As[buf][row1 * 32 + cc1]);
    async_load_b128(&Bw[(size_t)(bn + row0) * Kdim + k0 + cc0], &Bs[buf][row0 * 32 + cc0]);
    async_load_b128(&Bw[(size_t)(bn + row1) * Kdim + k0 + cc1], &Bs[buf][row1 * 32 + cc1]);
  };

  f32x8 acc[4][2];
#pragma unroll
  for (int mt = 0; mt < 4; ++mt)
#pragma unroll
    for (int nt = 0; nt < 2; ++nt) acc[mt][nt] = zero8();

  const int T = Kdim >> 5;             // k-tiles of 32
  issue(0, 0);
  for (int it = 0; it < T; ++it) {
    const int cur = it & 1;
    if (it + 1 < T) { issue(cur ^ 1, (it + 1) * 32); wait_async4(); }
    else            { wait_async0(); }
    __syncthreads();                   // tile `it` fully in LDS for all waves

    FragBF af[4], bf[2];
#pragma unroll
    for (int mt = 0; mt < 4; ++mt)
      load_afrag(af[mt], &As[cur][(wm + mt * 16 + lr) * 32], ahalf);
#pragma unroll
    for (int nt = 0; nt < 2; ++nt)
      load_bfrag(bf[nt], &Bs[cur][(wn + nt * 16 + lr) * 32], bhalf);
#pragma unroll
    for (int mt = 0; mt < 4; ++mt)     // 8 independent accumulation chains
#pragma unroll
      for (int nt = 0; nt < 2; ++nt)
        acc[mt][nt] = wmma_bf16(af[mt], bf[nt], acc[mt][nt]);

    __syncthreads();                   // readers done before buffer reuse
  }

  // epilogue: C layout -> VGPR r holds row (r + 8*(lane>=16)), col = lane&15
#pragma unroll
  for (int mt = 0; mt < 4; ++mt)
#pragma unroll
    for (int nt = 0; nt < 2; ++nt)
#pragma unroll
      for (int r = 0; r < 8; ++r) {
        const int m = bm + wm + mt * 16 + r + ((lane & 16) ? 8 : 0);
        const int n = bn + wn + nt * 16 + lr;
        float v = acc[mt][nt][r] * scale;
        if constexpr (MODE == 0) {
          ((unsigned short*)Cout)[(size_t)m * Ndim + n] = f2bf(v);
        } else if constexpr (MODE == 1) {
          const int s = m & (Sc - 1), bb = m >> 11;       // Sc = 2048
          const int h = n >> 6,      d  = n & 63;         // Dc = 64
          ((unsigned short*)Cout)[((size_t)(bb * Hc + h) * Dc + d) * Sc + s] = f2bf(v);
        } else if constexpr (MODE == 2) {
          const int s = m & (Sc - 1), bb = m >> 11;
          ((float*)Cout)[((size_t)bb * Ec + n) * Sc + s] = v + bias[n];
        } else {                                          // MODE 3
          float t = v + bias[n];
          t = fmaxf(t, 0.f);
          ((unsigned short*)Cout)[(size_t)m * Ndim + n] = f2bf(t);
        }
      }
}

// =====================================================================
// causal flash attention, one wave per 16 query rows, 64-key tiles
// (4 independent score chains -> good WMMA ILP, softmax VALU amortized)
// Q,K: bf16 [B,S,E] head slices; Vt: bf16 [B,H,D,S]; O: fp32 [B,S,E]
// =====================================================================
__global__ __launch_bounds__(128) void flash_attn_kernel(
    const unsigned short* __restrict__ Q,
    const unsigned short* __restrict__ Km,
    const unsigned short* __restrict__ Vt,
    float* __restrict__ O) {
  const int wave = threadIdx.x >> 5;
  const int lane = threadIdx.x & 31;
  const int b  = blockIdx.z;
  const int h  = blockIdx.y;
  const int qb = blockIdx.x * 64 + wave * 16;

  const unsigned short* qp = Q  + (size_t)b * Sc * Ec + h * Dc;
  const unsigned short* kp = Km + (size_t)b * Sc * Ec + h * Dc;
  const unsigned short* vp = Vt + (size_t)(b * Hc + h) * Dc * Sc;

  __shared__ __align__(16) unsigned short pbuf[4][16 * 64];
  unsigned short* pb = pbuf[wave];

  const int lr    = lane & 15;
  const int ahalf = (lane & 16) ? 8 : 0;
  const int bhalf = (lane & 16) ? 8 : 0;
  const int rhi   = (lane & 16) ? 8 : 0;

  FragBF aq[2];                       // Q rows qb..qb+15, K-dim D=64
#pragma unroll
  for (int kk = 0; kk < 2; ++kk)
    load_afrag(aq[kk], qp + (size_t)(qb + lr) * Ec + kk * 32, ahalf);

  float mrow[8], lrow[8];
  f32x8 oacc[4];
#pragma unroll
  for (int r = 0; r < 8; ++r) { mrow[r] = -1e30f; lrow[r] = 0.f; }
#pragma unroll
  for (int dt = 0; dt < 4; ++dt) oacc[dt] = zero8();

  for (int jb = 0; jb < qb + 16; jb += 64) {
    // ---- scores: four 16x16 tiles over 64 keys, K-dim 64 ----
    f32x8 sc[4];
#pragma unroll
    for (int t = 0; t < 4; ++t) sc[t] = zero8();
    FragBF kf;
#pragma unroll
    for (int kk = 0; kk < 2; ++kk)     // interleave 4 independent chains
#pragma unroll
      for (int t = 0; t < 4; ++t) {
        load_bfrag(kf, kp + (size_t)(jb + 16 * t + lr) * Ec + kk * 32, bhalf);
        sc[t] = wmma_bf16(aq[kk], kf, sc[t]);
      }

#pragma unroll
    for (int r = 0; r < 8; ++r) {
      const int qi = qb + r + rhi;
      float p[4];
      float mx = -1e30f;
#pragma unroll
      for (int t = 0; t < 4; ++t) {
        float a = sc[t][r];
        if (jb + 16 * t + 16 > qb) {          // tile touches the diagonal
          if (jb + 16 * t + lr > qi) a = -1e30f;
        }
        p[t] = a;
        mx = fmaxf(mx, a);
      }
      mx = fmaxf(mx, __shfl_xor(mx, 1, 16));
      mx = fmaxf(mx, __shfl_xor(mx, 2, 16));
      mx = fmaxf(mx, __shfl_xor(mx, 4, 16));
      mx = fmaxf(mx, __shfl_xor(mx, 8, 16));
      const float mn    = fmaxf(mrow[r], mx);
      const float alpha = __expf(mrow[r] - mn);
      float rs = 0.f;
#pragma unroll
      for (int t = 0; t < 4; ++t) { p[t] = __expf(p[t] - mn); rs += p[t]; }
      rs += __shfl_xor(rs, 1, 16);
      rs += __shfl_xor(rs, 2, 16);
      rs += __shfl_xor(rs, 4, 16);
      rs += __shfl_xor(rs, 8, 16);
      lrow[r] = lrow[r] * alpha + rs;
      mrow[r] = mn;
#pragma unroll
      for (int dt = 0; dt < 4; ++dt) oacc[dt][r] *= alpha;
      const int prow = r + rhi;                // P tile -> LDS [qrow, key]
#pragma unroll
      for (int t = 0; t < 4; ++t)
        pb[prow * 64 + 16 * t + lr] = f2bf(p[t]);
    }
    // wave-private LDS handoff (waves have different trip counts -> no barrier)
    asm volatile("s_wait_dscnt 0" ::: "memory");

    FragBF pf0, pf1;
    load_afrag(pf0, pb + lr * 64, ahalf);        // keys jb..jb+31
    load_afrag(pf1, pb + lr * 64 + 32, ahalf);   // keys jb+32..jb+63
    // ---- O += P(16x64) * V(64x64): two passes of 4 independent chains ----
#pragma unroll
    for (int dt = 0; dt < 4; ++dt) {
      FragBF vf;
      load_bfrag(vf, vp + (size_t)(dt * 16 + lr) * Sc + jb, bhalf);
      oacc[dt] = wmma_bf16(pf0, vf, oacc[dt]);
    }
#pragma unroll
    for (int dt = 0; dt < 4; ++dt) {
      FragBF vf;
      load_bfrag(vf, vp + (size_t)(dt * 16 + lr) * Sc + jb + 32, bhalf);
      oacc[dt] = wmma_bf16(pf1, vf, oacc[dt]);
    }
  }

  // normalize and scatter to [B,S,E]
#pragma unroll
  for (int dt = 0; dt < 4; ++dt) {
    const int col = h * Dc + dt * 16 + lr;
#pragma unroll
    for (int r = 0; r < 8; ++r) {
      const int row = qb + rhi + r;
      O[((size_t)b * Sc + row) * Ec + col] = oacc[dt][r] / lrow[r];
    }
  }
}

// =====================================================================
// combined = attn_l - lam*attn_r ; LayerNorm -> bf16, one block per row
// =====================================================================
__global__ __launch_bounds__(256) void combine_ln_kernel(
    const float* __restrict__ ol, const float* __restrict__ orr,
    const float* __restrict__ lql, const float* __restrict__ lkl,
    const float* __restrict__ lqr, const float* __restrict__ lkr,
    const float* __restrict__ g,   const float* __restrict__ bt,
    unsigned short* __restrict__ yln) {
  __shared__ float red[256];
  __shared__ float red2[256];
  const int row = blockIdx.x;
  const int tid = threadIdx.x;

  float sl = 0.f, sr = 0.f;                      // lambda (redundant, cheap)
#pragma unroll 8
  for (int d = 0; d < Dc; ++d) { sl += lql[d] * lkl[d]; sr += lqr[d] * lkr[d]; }
  const float lam = __expf(sl) - __expf(sr) + kLamInit;

  const size_t base = (size_t)row * Ec;
  float v[4], s = 0.f, sq = 0.f;
#pragma unroll
  for (int i = 0; i < 4; ++i) {
    const int e = tid + i * 256;
    v[i] = ol[base + e] - lam * orr[base + e];
    s  += v[i];
    sq += v[i] * v[i];
  }
  red[tid] = s; red2[tid] = sq;
  __syncthreads();
  for (int off = 128; off > 0; off >>= 1) {
    if (tid < off) { red[tid] += red[tid + off]; red2[tid] += red2[tid + off]; }
    __syncthreads();
  }
  const float mu  = red[0] * (1.f / Ec);
  const float var = red2[0] * (1.f / Ec) - mu * mu;
  const float rs  = rsqrtf(var + kLnEps);
#pragma unroll
  for (int i = 0; i < 4; ++i) {
    const int e = tid + i * 256;
    yln[base + e] = f2bf((v[i] - mu) * rs * g[e] + bt[e]);
  }
}

// =====================================================================
// host launcher
// =====================================================================
extern "C" void kernel_launch(void* const* d_in, const int* in_sizes, int n_in,
                              void* d_out, int out_size, void* d_ws, size_t ws_size,
                              hipStream_t stream) {
  const float* left  = (const float*)d_in[0];
  const float* right = (const float*)d_in[1];
  const float* Wq_l  = (const float*)d_in[2];
  const float* Wk_l  = (const float*)d_in[3];
  const float* Wv_l  = (const float*)d_in[4];
  const float* Wq_r  = (const float*)d_in[5];
  const float* Wk_r  = (const float*)d_in[6];
  const float* Wv_r  = (const float*)d_in[7];
  const float* Wo    = (const float*)d_in[8];
  const float* lql   = (const float*)d_in[9];
  const float* lkl   = (const float*)d_in[10];
  const float* lqr   = (const float*)d_in[11];
  const float* lkr   = (const float*)d_in[12];
  const float* ln_g  = (const float*)d_in[13];
  const float* ln_b  = (const float*)d_in[14];
  const float* W1    = (const float*)d_in[15];
  const float* b1    = (const float*)d_in[16];
  const float* W2    = (const float*)d_in[17];
  const float* b2    = (const float*)d_in[18];
  float* out = (float*)d_out;

  // ---- workspace layout ----
  size_t off = 0;
  auto alloc = [&](size_t nbytes) -> void* {
    void* p = (char*)d_ws + off;
    off = (off + nbytes + 255) & ~((size_t)255);
    return p;
  };
  const size_t MN = (size_t)Mc * Ec;                     // 4096*1024
  unsigned short* xl  = (unsigned short*)alloc(MN * 2);
  unsigned short* xr  = (unsigned short*)alloc(MN * 2);
  unsigned short* wqlb = (unsigned short*)alloc((size_t)Ec * Ec * 2);
  unsigned short* wklb = (unsigned short*)alloc((size_t)Ec * Ec * 2);
  unsigned short* wvlb = (unsigned short*)alloc((size_t)Ec * Ec * 2);
  unsigned short* wqrb = (unsigned short*)alloc((size_t)Ec * Ec * 2);
  unsigned short* wkrb = (unsigned short*)alloc((size_t)Ec * Ec * 2);
  unsigned short* wvrb = (unsigned short*)alloc((size_t)Ec * Ec * 2);
  unsigned short* wob  = (unsigned short*)alloc((size_t)Ec * Ec * 2);
  unsigned short* w1b  = (unsigned short*)alloc((size_t)2 * Ec * Ec * 2);
  unsigned short* w2b  = (unsigned short*)alloc((size_t)2 * Ec * Ec * 2);
  unsigned short* ql  = (unsigned short*)alloc(MN * 2);
  unsigned short* kl  = (unsigned short*)alloc(MN * 2);
  unsigned short* qr  = (unsigned short*)alloc(MN * 2);
  unsigned short* kr  = (unsigned short*)alloc(MN * 2);
  unsigned short* vtl = (unsigned short*)alloc(MN * 2);  // [B,H,D,S]
  unsigned short* vtr = (unsigned short*)alloc(MN * 2);
  float* ol  = (float*)alloc(MN * 4);
  float* orr = (float*)alloc(MN * 4);
  unsigned short* yln = (unsigned short*)alloc(MN * 2);
  unsigned short* yo  = (unsigned short*)alloc(MN * 2);
  unsigned short* h1  = (unsigned short*)alloc((size_t)Mc * 2 * Ec * 2);

  // ---- weight casts ----
  const int esz = Ec * Ec;
  cast_bf16_kernel<<<512, 256, 0, stream>>>(Wq_l, wqlb, esz);
  cast_bf16_kernel<<<512, 256, 0, stream>>>(Wk_l, wklb, esz);
  cast_bf16_kernel<<<512, 256, 0, stream>>>(Wv_l, wvlb, esz);
  cast_bf16_kernel<<<512, 256, 0, stream>>>(Wq_r, wqrb, esz);
  cast_bf16_kernel<<<512, 256, 0, stream>>>(Wk_r, wkrb, esz);
  cast_bf16_kernel<<<512, 256, 0, stream>>>(Wv_r, wvrb, esz);
  cast_bf16_kernel<<<512, 256, 0, stream>>>(Wo,   wob,  esz);
  cast_bf16_kernel<<<512, 256, 0, stream>>>(W1, w1b, 2 * esz);
  cast_bf16_kernel<<<512, 256, 0, stream>>>(W2, w2b, 2 * esz);

  // ---- input transpose+cast [B,E,S] -> [B,S,E] ----
  {
    dim3 g(Sc / 32, Ec / 32, Bc), blk(32, 8);
    tcast_kernel<<<g, blk, 0, stream>>>(left,  xl);
    tcast_kernel<<<g, blk, 0, stream>>>(right, xr);
  }

  // ---- QKV projections (q pre-scaled by D^-1/2) ----
  const float qscale = 0.125f;  // 64^-0.5
  {
    dim3 g(Ec / 128, Mc / 128);
    gemm_bf16_kernel<0><<<g, 256, 0, stream>>>(xl, wqlb, nullptr, ql,  Mc, Ec, Ec, qscale);
    gemm_bf16_kernel<0><<<g, 256, 0, stream>>>(xl, wklb, nullptr, kl,  Mc, Ec, Ec, 1.f);
    gemm_bf16_kernel<1><<<g, 256, 0, stream>>>(xl, wvlb, nullptr, vtl, Mc, Ec, Ec, 1.f);
    gemm_bf16_kernel<0><<<g, 256, 0, stream>>>(xr, wqrb, nullptr, qr,  Mc, Ec, Ec, qscale);
    gemm_bf16_kernel<0><<<g, 256, 0, stream>>>(xr, wkrb, nullptr, kr,  Mc, Ec, Ec, 1.f);
    gemm_bf16_kernel<1><<<g, 256, 0, stream>>>(xr, wvrb, nullptr, vtr, Mc, Ec, Ec, 1.f);
  }

  // ---- causal flash attention, both paths ----
  {
    dim3 g(Sc / 64, Hc, Bc);
    flash_attn_kernel<<<g, 128, 0, stream>>>(ql, kl, vtl, ol);
    flash_attn_kernel<<<g, 128, 0, stream>>>(qr, kr, vtr, orr);
  }

  // ---- differential combine + LayerNorm ----
  combine_ln_kernel<<<Mc, 256, 0, stream>>>(ol, orr, lql, lkl, lqr, lkr,
                                            ln_g, ln_b, yln);

  // ---- output projection + FFN ----
  {
    dim3 g(Ec / 128, Mc / 128);
    gemm_bf16_kernel<0><<<g, 256, 0, stream>>>(yln, wob, nullptr, yo, Mc, Ec, Ec, 1.f);
  }
  {
    dim3 g((2 * Ec) / 128, Mc / 128);
    gemm_bf16_kernel<3><<<g, 256, 0, stream>>>(yo, w1b, b1, h1, Mc, 2 * Ec, Ec, 1.f);
  }
  {
    dim3 g(Ec / 128, Mc / 128);   // final: fp32, +b2, transposed to [B,E,S]
    gemm_bf16_kernel<2><<<g, 256, 0, stream>>>(h1, w2b, b2, out, Mc, Ec, 2 * Ec, 1.f);
  }
  (void)in_sizes; (void)n_in; (void)out_size; (void)ws_size;
}
